// BidirectionalLSTM_50130858279291
// MI455X (gfx1250) — matrix operand order repaired and checked by reading, not compiled, not run on hardware
//
#include <hip/hip_runtime.h>
#include <hip/hip_bf16.h>

#define B_ 512
#define S_ 512
#define V_ 128
#define E_ 128
#define H_ 256
#define NG 1024          // 4*H gate width
#define MROWS 32         // batch rows per block
#define NBLK (B_ / MROWS)
#define TPB 256          // 8 waves (wave32)

typedef __bf16 bf16_t;
typedef bf16_t v16bf __attribute__((ext_vector_type(16)));
typedef float  v8f   __attribute__((ext_vector_type(8)));

__device__ __forceinline__ unsigned short f2bf(float x) {
  unsigned int u = __builtin_bit_cast(unsigned int, x);
  u += 0x7FFFu + ((u >> 16) & 1u);          // round-to-nearest-even
  return (unsigned short)(u >> 16);
}
__device__ __forceinline__ float sigm(float x) { return 1.0f / (1.0f + __expf(-x)); }
__device__ __forceinline__ float tanh_f(float x) {
  float e = __expf(2.0f * x);
  return (e - 1.0f) / (e + 1.0f);
}

// ---------------- weight repack kernels (run once per launch, tiny) ----------------

// Wt[k][n] (bf16, row pitch NG): k<Din -> Wi[n][k] ; k>=Din -> Wh[n][k-Din].
// Also combined bias[n] = bi[n] + bh[n].
__global__ void repack_w(const float* __restrict__ Wi, const float* __restrict__ Wh,
                         const float* __restrict__ bi, const float* __restrict__ bh,
                         unsigned short* __restrict__ Wt, float* __restrict__ bias, int Din) {
  int KC = Din + H_;
  int total = KC * NG;
  for (int idx = blockIdx.x * blockDim.x + threadIdx.x; idx < total;
       idx += gridDim.x * blockDim.x) {
    int k = idx / NG, n = idx % NG;
    float v = (k < Din) ? Wi[n * Din + k] : Wh[n * H_ + (k - Din)];
    Wt[idx] = f2bf(v);
  }
  for (int n = blockIdx.x * blockDim.x + threadIdx.x; n < NG; n += gridDim.x * blockDim.x)
    bias[n] = bi[n] + bh[n];
}

// Wgt[k][n] = Wg[n][k] (256x128 bf16); embB = bf16(emb) (128x128).
__global__ void repack_misc(const float* __restrict__ Wg, unsigned short* __restrict__ Wgt,
                            const float* __restrict__ emb, unsigned short* __restrict__ embB) {
  int idx = blockIdx.x * blockDim.x + threadIdx.x;
  if (idx < H_ * V_) {
    int k = idx / V_, n = idx % V_;
    Wgt[idx] = f2bf(Wg[n * H_ + k]);
  }
  if (idx < V_ * E_) embB[idx] = f2bf(emb[idx]);
}

// ---------------- fused LSTM layer kernel ----------------
// Each block owns 32 batch rows and runs all S timesteps independently.
// A (LDS, bf16) = [ x_t (DIN) | h_{t-1} (H) ] per row; weights streamed from L2 as
// pre-transposed bf16 [KC x 1024]; gates accumulated in f32 WMMA, cell update in LDS.
// LAYER1 additionally fuses the V=128 vocab projection per timestep.
template <int DIN, bool LAYER1>
__global__ __launch_bounds__(TPB) void lstm_kernel(
    const long long* __restrict__ xidx,        // layer0: [B,S] token ids
    const unsigned short* __restrict__ embB,   // layer0: bf16 emb [V,E]
    const unsigned short* __restrict__ in_seq, // layer1: bf16 h1 [B,S,H]
    const unsigned short* __restrict__ Wt,     // bf16 [DIN+H, 1024]
    const float* __restrict__ bias,            // f32 [1024] (bi+bh)
    const unsigned short* __restrict__ Wgt,    // layer1: bf16 [H, V]
    const float* __restrict__ bg,              // layer1: f32 [V]
    unsigned short* __restrict__ out_seq,      // layer0: bf16 h1 [B,S,H]
    float* __restrict__ logits)                // layer1: f32 [B,S,V]
{
  constexpr int KC = DIN + H_;
  constexpr int APITCH = KC + 8;   // mult of 8 -> 16B-aligned frag loads, bank skew
  constexpr int GPITCH = NG + 4;

  extern __shared__ char smem[];
  unsigned short* A = (unsigned short*)smem;                                 // 32 x APITCH bf16
  float* gatesL = (float*)(smem + (size_t)MROWS * APITCH * 2);               // 32 x GPITCH f32
  float* cL = (float*)(smem + (size_t)MROWS * APITCH * 2 + (size_t)MROWS * GPITCH * 4); // 32 x 256

  const int tid = threadIdx.x;
  const int wave = tid >> 5;
  const int lane = tid & 31;
  const int b0 = blockIdx.x * MROWS;

  const int mloc = lane & 15;            // A-frag row within 16-row tile
  const int khalf = (lane >> 4) << 3;    // A-frag k-phase: 0 or 8
  const int ccol = lane & 15;            // C/D: N = lane&15
  const int crow = (lane >> 4) << 3;     // C/D: M = r + 8*(lane>>4)

  // zero h-region of A and cell state
  for (int idx = tid; idx < MROWS * H_; idx += TPB) {
    int m = idx >> 8, j = idx & (H_ - 1);
    A[m * APITCH + DIN + j] = 0;
    cL[idx] = 0.0f;
  }
  __syncthreads();

  for (int t = 0; t < S_; ++t) {
    // ---- phase 1: stage x_t into A[:, 0:DIN] ----
    if (!LAYER1) {
      for (int idx = tid; idx < MROWS * E_; idx += TPB) {
        int m = idx >> 7, e = idx & (E_ - 1);
        long long v = xidx[(long long)(b0 + m) * S_ + t];
        A[m * APITCH + e] = embB[(int)v * E_ + e];
      }
    } else {
      for (int idx = tid; idx < MROWS * (H_ / 2); idx += TPB) {
        int m = idx >> 7, d2 = idx & 127;
        const unsigned int* src =
            (const unsigned int*)(in_seq + (((long long)(b0 + m)) * S_ + t) * H_);
        *(unsigned int*)(A + m * APITCH + d2 * 2) = src[d2];
      }
    }
    __syncthreads();

    // ---- phase 2: gates = [x_t | h] @ Wt  (bf16 WMMA, f32 acc) ----
    v8f acc[2][8];
#pragma unroll
    for (int rt = 0; rt < 2; ++rt)
#pragma unroll
      for (int nt = 0; nt < 8; ++nt) acc[rt][nt] = v8f{};

    for (int kc = 0; kc < KC; kc += 32) {
      v16bf af[2];
#pragma unroll
      for (int rt = 0; rt < 2; ++rt) {
        const unsigned short* ap = A + (rt * 16 + mloc) * APITCH + kc + khalf;
        union { v16bf v; uint4 q[2]; } u;
        u.q[0] = *(const uint4*)(ap);        // K = kc+khalf .. +7
        u.q[1] = *(const uint4*)(ap + 16);   // K = kc+16+khalf .. +7
        af[rt] = u.v;
      }
#pragma unroll
      for (int nt = 0; nt < 8; ++nt) {
        const int n0 = (wave << 7) + (nt << 4);
        const unsigned short* bp = Wt + (size_t)(kc + khalf) * NG + n0 + ccol; // N = n0+(lane&15)
        union { v16bf v; unsigned short s[16]; } ub;
#pragma unroll
        for (int i = 0; i < 8; ++i) {
          ub.s[i] = bp[(size_t)i * NG];             // K = kc+khalf+i
          ub.s[8 + i] = bp[(size_t)(16 + i) * NG];  // K = kc+16+khalf+i
        }
        acc[0][nt] = __builtin_amdgcn_wmma_f32_16x16x32_bf16(
            false, af[0], false, ub.v, (short)0, acc[0][nt], false, false);
        acc[1][nt] = __builtin_amdgcn_wmma_f32_16x16x32_bf16(
            false, af[1], false, ub.v, (short)0, acc[1][nt], false, false);
      }
    }

#pragma unroll
    for (int rt = 0; rt < 2; ++rt)
#pragma unroll
      for (int nt = 0; nt < 8; ++nt) {
        const int n = (wave << 7) + (nt << 4) + ccol;
#pragma unroll
        for (int r = 0; r < 8; ++r) {
          const int m = rt * 16 + crow + r;
          gatesL[m * GPITCH + n] = acc[rt][nt][r];
        }
      }
    __syncthreads();

    // ---- phase 3: LSTM cell update (gate order f,i,g,o) ----
    for (int idx = tid; idx < MROWS * H_; idx += TPB) {
      int m = idx >> 8, j = idx & (H_ - 1);
      const float* gr = gatesL + m * GPITCH;
      float fp = gr[j] + bias[j];
      float ip = gr[H_ + j] + bias[H_ + j];
      float gp = gr[2 * H_ + j] + bias[2 * H_ + j];
      float op = gr[3 * H_ + j] + bias[3 * H_ + j];
      float c = sigm(fp) * cL[idx] + sigm(ip) * tanh_f(gp);
      cL[idx] = c;
      float h = sigm(op) * tanh_f(c);
      unsigned short hb = f2bf(h);
      A[m * APITCH + DIN + j] = hb;
      if (!LAYER1) out_seq[(((long long)(b0 + m)) * S_ + t) * H_ + j] = hb;
    }
    __syncthreads();

    // ---- phase 4 (layer1): fused vocab projection logits_t = h2_t @ Wgt + bg ----
    if (LAYER1) {
      v8f lacc[2];
      lacc[0] = v8f{};
      lacc[1] = v8f{};
      for (int kc = 0; kc < H_; kc += 32) {
        v16bf af[2];
#pragma unroll
        for (int rt = 0; rt < 2; ++rt) {
          const unsigned short* ap = A + (rt * 16 + mloc) * APITCH + DIN + kc + khalf;
          union { v16bf v; uint4 q[2]; } u;
          u.q[0] = *(const uint4*)(ap);
          u.q[1] = *(const uint4*)(ap + 16);
          af[rt] = u.v;
        }
        const unsigned short* bp = Wgt + (kc + khalf) * V_ + (wave << 4) + ccol;
        union { v16bf v; unsigned short s[16]; } ub;
#pragma unroll
        for (int i = 0; i < 8; ++i) {
          ub.s[i] = bp[i * V_];
          ub.s[8 + i] = bp[(16 + i) * V_];
        }
        lacc[0] = __builtin_amdgcn_wmma_f32_16x16x32_bf16(
            false, af[0], false, ub.v, (short)0, lacc[0], false, false);
        lacc[1] = __builtin_amdgcn_wmma_f32_16x16x32_bf16(
            false, af[1], false, ub.v, (short)0, lacc[1], false, false);
      }
      const int n = (wave << 4) + ccol;
      const float bgv = bg[n];
#pragma unroll
      for (int rt = 0; rt < 2; ++rt)
#pragma unroll
        for (int r = 0; r < 8; ++r) {
          const int m = rt * 16 + crow + r;
          logits[(((long long)(b0 + m)) * S_ + t) * V_ + n] = lacc[rt][r] + bgv;
        }
    }
  }
}

extern "C" void kernel_launch(void* const* d_in, const int* in_sizes, int n_in,
                              void* d_out, int out_size, void* d_ws, size_t ws_size,
                              hipStream_t stream) {
  (void)in_sizes; (void)n_in; (void)out_size; (void)ws_size;
  const long long* x = (const long long*)d_in[0];
  const float* emb = (const float*)d_in[1];
  const float* Wi0 = (const float*)d_in[2];
  const float* bi0 = (const float*)d_in[3];
  const float* Wh0 = (const float*)d_in[4];
  const float* bh0 = (const float*)d_in[5];
  const float* Wi1 = (const float*)d_in[6];
  const float* bi1 = (const float*)d_in[7];
  const float* Wh1 = (const float*)d_in[8];
  const float* bh1 = (const float*)d_in[9];
  const float* Wg = (const float*)d_in[10];
  const float* bg = (const float*)d_in[11];

  char* ws = (char*)d_ws;
  size_t off = 0;
  auto alloc = [&](size_t bytes) {
    void* p = ws + off;
    off = (off + bytes + 255) & ~size_t(255);
    return p;
  };
  unsigned short* Wt0 = (unsigned short*)alloc((size_t)(E_ + H_) * NG * 2);
  unsigned short* Wt1 = (unsigned short*)alloc((size_t)(H_ + H_) * NG * 2);
  unsigned short* Wgt = (unsigned short*)alloc((size_t)H_ * V_ * 2);
  float* bias0 = (float*)alloc(NG * 4);
  float* bias1 = (float*)alloc(NG * 4);
  unsigned short* embB = (unsigned short*)alloc((size_t)V_ * E_ * 2);
  unsigned short* h1seq = (unsigned short*)alloc((size_t)B_ * S_ * H_ * 2);

  repack_w<<<512, 256, 0, stream>>>(Wi0, Wh0, bi0, bh0, Wt0, bias0, E_);
  repack_w<<<512, 256, 0, stream>>>(Wi1, Wh1, bi1, bh1, Wt1, bias1, H_);
  repack_misc<<<(H_ * V_ + 255) / 256, 256, 0, stream>>>(Wg, Wgt, emb, embB);

  const size_t smem0 = (size_t)MROWS * (E_ + H_ + 8) * 2 + (size_t)MROWS * (NG + 4) * 4 +
                       (size_t)MROWS * H_ * 4;  // ~189 KB
  const size_t smem1 = (size_t)MROWS * (H_ + H_ + 8) * 2 + (size_t)MROWS * (NG + 4) * 4 +
                       (size_t)MROWS * H_ * 4;  // ~197 KB (<320 KB/WGP)
  hipFuncSetAttribute((const void*)lstm_kernel<E_, false>,
                      hipFuncAttributeMaxDynamicSharedMemorySize, (int)smem0);
  hipFuncSetAttribute((const void*)lstm_kernel<H_, true>,
                      hipFuncAttributeMaxDynamicSharedMemorySize, (int)smem1);

  lstm_kernel<E_, false><<<NBLK, TPB, smem0, stream>>>(
      x, embB, nullptr, Wt0, bias0, nullptr, nullptr, h1seq, nullptr);
  lstm_kernel<H_, true><<<NBLK, TPB, smem1, stream>>>(
      nullptr, nullptr, h1seq, Wt1, bias1, Wgt, bg, nullptr, (float*)d_out);
}
